// SelfAttention_71614284694161
// MI455X (gfx1250) — matrix operand order repaired and checked by reading, compiled, not run on hardware
//
#include <hip/hip_runtime.h>
#include <hip/hip_bf16.h>

// ---------------------------------------------------------------------------
// Problem constants (reference: B=2, T=2048, C=1024, H=16, D=64)
// ---------------------------------------------------------------------------
#define BB  2
#define TT  2048
#define CC  1024
#define HH  16
#define DD  64

typedef __attribute__((ext_vector_type(16))) _Float16 v16h;
typedef __attribute__((ext_vector_type(8)))  _Float16 v8h;
typedef __attribute__((ext_vector_type(4)))  _Float16 v4h;
typedef __attribute__((ext_vector_type(8)))  float    v8f;
typedef __attribute__((ext_vector_type(4)))  float    v4f;

union HFrag { v16h v; v8h h8[2]; _Float16 h[16]; };

// A fragment (16x32 f16, MxK): lane<16 holds row M=lane, K = {k0..k0+7, k0+16..k0+23};
// lane>=16 holds row M=lane-16, K shifted by +8.  Caller passes p = &A[row*lda + k0 + ((lane>>4)<<3)].
__device__ __forceinline__ v16h load_afrag(const _Float16* p) {
  HFrag f;
  f.h8[0] = *(const v8h*)p;
  f.h8[1] = *(const v8h*)(p + 16);
  return f.v;
}
// B fragment (32x16 f16, KxN) from Bt[N,K] row-major: lane<16 holds col N=lane, K = {k0..k0+7, k0+16..k0+23};
// lane>=16 holds col N=lane-16, K shifted by +8.  Caller passes p = &Bt[col*ldb + k0 + ((lane>>4)<<3)].
__device__ __forceinline__ v16h load_bfrag(const _Float16* p) {
  HFrag f;
  f.h8[0] = *(const v8h*)p;
  f.h8[1] = *(const v8h*)(p + 16);
  return f.v;
}
__device__ __forceinline__ v8f wmma16(v16h a, v16h b, v8f c) {
  return __builtin_amdgcn_wmma_f32_16x16x32_f16(false, a, false, b, (short)0, c, false, false);
}

// ---------------------------------------------------------------------------
// 1a. pack f32 -> f16 (elementwise, 4 per thread)
// ---------------------------------------------------------------------------
__global__ void __launch_bounds__(256)
pack_f16_kernel(const float* __restrict__ in, _Float16* __restrict__ out, int n4) {
  int i = blockIdx.x * 256 + threadIdx.x;
  if (i >= n4) return;
  v4f v = *(const v4f*)(in + (size_t)i * 4);
  v4h o;
  o[0] = (_Float16)v[0]; o[1] = (_Float16)v[1];
  o[2] = (_Float16)v[2]; o[3] = (_Float16)v[3];
  *(v4h*)(out + (size_t)i * 4) = o;
}

// ---------------------------------------------------------------------------
// 1b. pack w_qkv -> f16, plus transposed f16 copy (for y @ w_qkv)
// ---------------------------------------------------------------------------
__global__ void __launch_bounds__(256)
pack_w_kernel(const float* __restrict__ w, _Float16* __restrict__ w16,
              _Float16* __restrict__ wT16) {
  int c = blockIdx.x * 256 + threadIdx.x;   // column
  int r = blockIdx.y;                       // row
  float v = w[(size_t)r * CC + c];
  w16[(size_t)r * CC + c]  = (_Float16)v;   // row-major (B^T for x @ W^T)
  wT16[(size_t)c * CC + r] = (_Float16)v;   // W^T row-major (B^T for y @ W)
}

// ---------------------------------------------------------------------------
// 2/4/7. Batched WMMA GEMM: C[M,N] = alpha * A[M,K] * Bt[N,K]^T   (f16 in, f32 out)
// One wave computes a 32x32 tile (2x2 WMMA frags, f32 accum). M,N,K mult. of 32.
// ---------------------------------------------------------------------------
__global__ void __launch_bounds__(128)
gemm_f16f16_f32(const _Float16* __restrict__ A, const _Float16* __restrict__ Bt,
                float* __restrict__ C, int M, int N, int K,
                long long sA, long long sB, long long sC, float alpha) {
  const int lane   = threadIdx.x & 31;
  const int tilesN = N >> 5;
  const int tile   = blockIdx.x * 4 + (threadIdx.x >> 5);
  if (tile >= (M >> 5) * tilesN) return;
  const int m0 = (tile / tilesN) << 5;
  const int n0 = (tile % tilesN) << 5;
  A  += (size_t)blockIdx.y * sA;
  Bt += (size_t)blockIdx.y * sB;
  C  += (size_t)blockIdx.y * sC;

  const _Float16* pa0 = A  + (size_t)(m0 + (lane & 15)) * K + ((lane >> 4) << 3);
  const _Float16* pa1 = pa0 + (size_t)16 * K;
  const _Float16* pb0 = Bt + (size_t)(n0 + (lane & 15)) * K + ((lane >> 4) << 3);
  const _Float16* pb1 = pb0 + (size_t)16 * K;

  v8f c00 = {}, c01 = {}, c10 = {}, c11 = {};
  for (int k0 = 0; k0 < K; k0 += 32) {
    v16h a0 = load_afrag(pa0 + k0);
    v16h a1 = load_afrag(pa1 + k0);
    v16h b0 = load_bfrag(pb0 + k0);
    v16h b1 = load_bfrag(pb1 + k0);
    __builtin_prefetch(pa0 + k0 + 512, 0, 1);   // global_prefetch_b8 next A tiles
    c00 = wmma16(a0, b0, c00);
    c01 = wmma16(a0, b1, c01);
    c10 = wmma16(a1, b0, c10);
    c11 = wmma16(a1, b1, c11);
  }
  // C/D layout: lane<16 -> (M = v, N = lane); lane>=16 -> (M = 8+v, N = lane-16)
  const int cn = n0 + (lane & 15);
  const int ml = (lane >> 4) << 3;
#pragma unroll
  for (int v = 0; v < 8; ++v) {
    size_t r0 = (size_t)(m0 + ml + v) * N;
    size_t r1 = (size_t)(m0 + 16 + ml + v) * N;
    C[r0 + cn]      = alpha * c00[v];
    C[r0 + cn + 16] = alpha * c01[v];
    C[r1 + cn]      = alpha * c10[v];
    C[r1 + cn + 16] = alpha * c11[v];
  }
}

// ---------------------------------------------------------------------------
// 3. Per-(b,t,h) RMSNorm over D=64; one wave32 per row; emits wn[B,H,T,D] f16
//    and wnT[B,H,D,T] f16 (K-contiguous B-operand for the att @ V GEMM).
//    Note: after this, each head-row has L2 norm == 8 (RMS 1 over 64 elems),
//    so scores a = (w_i . w_j)/8 satisfy |a| <= 8  ->  softmax needs NO max.
// ---------------------------------------------------------------------------
__global__ void __launch_bounds__(256)
rmsnorm_pack_kernel(const float* __restrict__ qkv, const float* __restrict__ g,
                    _Float16* __restrict__ wn, _Float16* __restrict__ wnT) {
  const int wid  = (int)((blockIdx.x * blockDim.x + threadIdx.x) >> 5); // (b,h,t) row
  const int lane = threadIdx.x & 31;
  if (wid >= BB * HH * TT) return;
  const int t = wid % TT;
  const int h = (wid / TT) % HH;
  const int b = wid / (TT * HH);

  const float* src = qkv + ((size_t)(b * TT + t)) * CC + h * DD;
  const int d0 = lane * 2;
  float2 x2 = *(const float2*)(src + d0);
  float ss = x2.x * x2.x + x2.y * x2.y;
#pragma unroll
  for (int off = 16; off > 0; off >>= 1) ss += __shfl_xor(ss, off, 32);
  const float r = rsqrtf(ss * (1.0f / (float)DD) + 1.1920929e-07f);
  const float o0 = x2.x * r * g[d0];
  const float o1 = x2.y * r * g[d0 + 1];

  _Float16* wrow = wn + (size_t)wid * DD;       // wn row index == (b*H+h)*T + t
  wrow[d0]     = (_Float16)o0;
  wrow[d0 + 1] = (_Float16)o1;
  const size_t tb = ((size_t)(b * HH + h)) * DD;
  wnT[(tb + d0) * TT + t]     = (_Float16)o0;
  wnT[(tb + d0 + 1) * TT + t] = (_Float16)o1;
}

// ---------------------------------------------------------------------------
// 5. Column softmax sums only (|a|<=8 -> no max pass): cinv[j] = 1/sum_i e^a.
//    Thread-per-column, fully coalesced scan down the rows, 1 exp/element.
// ---------------------------------------------------------------------------
__global__ void __launch_bounds__(256)
col_sum_kernel(const float* __restrict__ att, float* __restrict__ cinv) {
  const size_t bh = blockIdx.y;
  const int j = blockIdx.x * 256 + threadIdx.x;
  const float* p = att + bh * (size_t)TT * TT + j;
  float s0 = 0.f, s1 = 0.f;                 // 2-way for a little more ILP
  for (int i = 0; i < TT; i += 2) {
    s0 += __expf(p[(size_t)i * TT]);
    s1 += __expf(p[(size_t)(i + 1) * TT]);
  }
  cinv[bh * TT + j] = 1.0f / (s0 + s1);
}

// ---------------------------------------------------------------------------
// 6. Fused dual-softmax + att @ V, flash-style row normalization:
//      t       = exp(a_ij)                      (single exp, no max needed)
//      yR[i,:] += t * w_j        (row-softmax part, normalized at epilogue)
//      yC[i,:] += t*cinv_j * w_j (col-softmax part, cinv precomputed)
//      rowsum  += t
//      y = yR/rowsum + yC
//    One wave per 16 rows x full D=64: 2x4 WMMA accumulator sets.
//    This removes the entire row-stats pass (saves a 537MB att_raw read).
// ---------------------------------------------------------------------------
__global__ void __launch_bounds__(128)
attn_y_kernel(const float* __restrict__ att, const _Float16* __restrict__ wnT,
              const float* __restrict__ cinv, _Float16* __restrict__ yh) {
  const int bh   = blockIdx.y;
  const int b    = bh >> 4, h = bh & 15;
  const int lane = threadIdx.x & 31;
  const int tile = blockIdx.x * 4 + (threadIdx.x >> 5); // 128 tiles of 16 rows
  const int i0   = tile * 16;

  const float* attp = att + (size_t)bh * TT * TT;
  const float* ci   = cinv + (size_t)bh * TT;
  const _Float16* wb = wnT + ((size_t)bh * DD + (lane & 15)) * TT;

  const int r0 = i0 + (lane & 15);                 // A-frag row owned by this lane
  const float* prow = attp + (size_t)r0 * TT;
  const int klo = (lane >> 4) << 3;                // A-frag K sub-offset (0 or 8)
  const int kbo = (lane >> 4) << 3;                // B-frag K sub-offset (0 or 8)

  v8f accR[4] = {}, accY[4] = {};
  float rsum = 0.f;
  for (int j0 = 0; j0 < TT; j0 += 32) {
    const int kb = j0 + klo;
    float av[16], civ[16];
    *(v4f*)&av[0]   = *(const v4f*)(prow + kb);
    *(v4f*)&av[4]   = *(const v4f*)(prow + kb + 4);
    *(v4f*)&av[8]   = *(const v4f*)(prow + kb + 16);
    *(v4f*)&av[12]  = *(const v4f*)(prow + kb + 20);
    *(v4f*)&civ[0]  = *(const v4f*)(ci + kb);
    *(v4f*)&civ[4]  = *(const v4f*)(ci + kb + 4);
    *(v4f*)&civ[8]  = *(const v4f*)(ci + kb + 16);
    *(v4f*)&civ[12] = *(const v4f*)(ci + kb + 20);
    __builtin_prefetch(prow + kb + 64, 0, 1);      // next att chunk of this row

    HFrag aR, aC;
#pragma unroll
    for (int t = 0; t < 16; ++t) {
      float e = __expf(av[t]);                     // |av|<=8: safe without max
      rsum += e;
      aR.h[t] = (_Float16)e;
      aC.h[t] = (_Float16)(e * civ[t]);
    }
    const _Float16* pb = wb + j0 + kbo;
    v16h b0 = load_bfrag(pb);
    v16h b1 = load_bfrag(pb + (size_t)16 * TT);
    v16h b2 = load_bfrag(pb + (size_t)32 * TT);
    v16h b3 = load_bfrag(pb + (size_t)48 * TT);
    accR[0] = wmma16(aR.v, b0, accR[0]);
    accR[1] = wmma16(aR.v, b1, accR[1]);
    accR[2] = wmma16(aR.v, b2, accR[2]);
    accR[3] = wmma16(aR.v, b3, accR[3]);
    accY[0] = wmma16(aC.v, b0, accY[0]);
    accY[1] = wmma16(aC.v, b1, accY[1]);
    accY[2] = wmma16(aC.v, b2, accY[2]);
    accY[3] = wmma16(aC.v, b3, accY[3]);
  }
  // lane L and L+16 hold complementary K-halves of row i0+(L&15): combine.
  rsum += __shfl_xor(rsum, 16, 32);
  const float rinv_own = 1.0f / rsum;              // full rowsum of row i0+(lane&15)

  // C/D layout: lane<16 -> (M=ml+v, N=lane); lane>=16 -> (M=ml+v, N=lane-16), ml=8
  const int ml = (lane >> 4) << 3;
  const int nl = lane & 15;
#pragma unroll
  for (int v = 0; v < 8; ++v) {
    const float rinv_v = __shfl(rinv_own, ml + v, 32);  // rowsum of row i0+ml+v
    const int mrow = i0 + ml + v;
#pragma unroll
    for (int nf = 0; nf < 4; ++nf) {
      const int d = nf * 16 + nl;
      yh[((size_t)(b * TT + mrow)) * CC + h * DD + d] =
          (_Float16)(accR[nf][v] * rinv_v + accY[nf][v]);
    }
  }
}

// ---------------------------------------------------------------------------
// Launch sequence
// ---------------------------------------------------------------------------
extern "C" void kernel_launch(void* const* d_in, const int* in_sizes, int n_in,
                              void* d_out, int out_size, void* d_ws, size_t ws_size,
                              hipStream_t stream) {
  (void)in_sizes; (void)n_in; (void)out_size; (void)ws_size;
  const size_t BT  = (size_t)BB * TT;
  const size_t BTC = BT * CC;
  const size_t BHT = (size_t)BB * HH * TT;

  const float* x  = (const float*)d_in[0];
  const float* wq = (const float*)d_in[1];
  const float* g  = (const float*)d_in[2];

  float* y_out = (float*)d_out;         // [B,T,C]
  float* att   = (float*)d_out + BTC;   // [B,H,T,T] (att_raw output, also scratch)

  char* wp = (char*)d_ws;
  auto alloc = [&](size_t bytes) -> char* {
    char* r = wp;
    wp += (bytes + 255) & ~(size_t)255;
    return r;
  };
  float*    qkv   = (float*)alloc(BTC * 4);                    // 16.8 MB
  _Float16* x16   = (_Float16*)alloc(BTC * 2);                 //  8.4 MB
  _Float16* wq16  = (_Float16*)alloc((size_t)CC * CC * 2);     //  2.1 MB
  _Float16* wqT16 = (_Float16*)alloc((size_t)CC * CC * 2);     //  2.1 MB
  _Float16* wn16  = (_Float16*)alloc(BHT * DD * 2);            //  8.4 MB
  _Float16* wnT16 = (_Float16*)alloc(BHT * DD * 2);            //  8.4 MB
  _Float16* yh16  = (_Float16*)alloc(BTC * 2);                 //  8.4 MB
  float*    cinvb = (float*)alloc(BHT * 4);                    //  0.3 MB

  // 1. precision packs
  pack_f16_kernel<<<dim3((unsigned)(BTC / 4 / 256)), 256, 0, stream>>>(x, x16, (int)(BTC / 4));
  pack_w_kernel<<<dim3(CC / 256, CC), 256, 0, stream>>>(wq, wq16, wqT16);

  // 2. qkv = x @ W^T   (M=4096, N=1024, K=1024)
  {
    int tiles = (int)(BT / 32) * (CC / 32);
    gemm_f16f16_f32<<<dim3((tiles + 3) / 4, 1), 128, 0, stream>>>(
        x16, wq16, qkv, (int)BT, CC, CC, 0, 0, 0, 1.0f);
  }

  // 3. per-head RMSNorm -> wn (f16) and wn^T (f16)
  rmsnorm_pack_kernel<<<dim3((unsigned)(BHT / 8)), 256, 0, stream>>>(qkv, g, wn16, wnT16);

  // 4. att_raw = scale * wn @ wn^T per (b,h)   (M=N=2048, K=64, batch=32)
  {
    int tiles = (TT / 32) * (TT / 32);
    gemm_f16f16_f32<<<dim3((tiles + 3) / 4, BB * HH), 128, 0, stream>>>(
        wn16, wn16, att, TT, TT, DD,
        (long long)TT * DD, (long long)TT * DD, (long long)TT * TT, 0.125f);
  }

  // 5. column softmax sums (row sums are computed on the fly in step 6)
  col_sum_kernel<<<dim3(TT / 256, BB * HH), 256, 0, stream>>>(att, cinvb);

  // 6. y_heads = (softmax_row + softmax_col)(att_raw) @ wn, fused flash-style
  attn_y_kernel<<<dim3(TT / 16 / 4, BB * HH), 128, 0, stream>>>(att, wnT16, cinvb, yh16);

  // 7. y = y_heads @ W   (M=4096, N=1024, K=1024)
  {
    int tiles = (int)(BT / 32) * (CC / 32);
    gemm_f16f16_f32<<<dim3((tiles + 3) / 4, 1), 128, 0, stream>>>(
        yh16, wqT16, y_out, (int)BT, CC, CC, 0, 0, 0, 1.0f);
  }
}
